// FHEReducedBitGPT_7679401525972
// MI455X (gfx1250) — compile-verified, matrix operand order checked by reading, not run on hardware
//
#include <hip/hip_runtime.h>

// ---------------------------------------------------------------------------
// Types
// ---------------------------------------------------------------------------
typedef __attribute__((ext_vector_type(16))) _Float16 v16h;
typedef __attribute__((ext_vector_type(8)))  float    v8f;
typedef __attribute__((ext_vector_type(4)))  unsigned int u32x4;
typedef int v4i_b128 __attribute__((vector_size(16)));   // builtin's pointee type

#define BM 128
#define BN 128
#define BK 32
#define LDSS 40   // halves per LDS row: 80 bytes (multiple of 16B)

union Frag {
    v16h     v;
    u32x4    q[2];
    _Float16 h[16];
};

// ---------------------------------------------------------------------------
// CDNA5 async global->LDS copy (16 bytes per lane), ASYNCcnt-tracked.
// Builtin signature (from hipcc diagnostic): (v4i global*, v4i shared*, off, cpol)
// ---------------------------------------------------------------------------
__device__ __forceinline__ void async_cp16B(const void* g, void* l)
{
#if __has_builtin(__builtin_amdgcn_global_load_async_to_lds_b128)
    __builtin_amdgcn_global_load_async_to_lds_b128(
        (__attribute__((address_space(1))) v4i_b128*)g,
        (__attribute__((address_space(3))) v4i_b128*)l, 0, 0);
#else
    unsigned ldsoff = (unsigned)(unsigned long long)
        (__attribute__((address_space(3))) void*)l;
    asm volatile("global_load_async_to_lds_b128 %0, %1, off"
                 :: "v"(ldsoff), "v"((unsigned long long)g) : "memory");
#endif
}

__device__ __forceinline__ void wait_async0()
{
#if __has_builtin(__builtin_amdgcn_s_wait_asynccnt)
    __builtin_amdgcn_s_wait_asynccnt(0);
#else
    asm volatile("s_wait_asynccnt 0x0" ::: "memory");
#endif
}

// ---------------------------------------------------------------------------
// Tile staging into LDS buffer (A always f16 via async DMA; B per layout)
//   BLAYOUT 0: B f32, [N,K] K-contig  -> load f32, cvt f16 in VGPR, ds_store
//   BLAYOUT 1: B f16, [N,K] K-contig  -> async DMA straight to LDS
//   BLAYOUT 2: B f16, [K,N] N-contig  -> load, transpose-scatter to LDS
// ---------------------------------------------------------------------------
template<int BLAYOUT>
__device__ __forceinline__ void stage_tile(
    const _Float16* __restrict__ Ab, const float* __restrict__ Bf,
    const _Float16* __restrict__ Bh, _Float16* lA, _Float16* lB,
    int bm, int bn, int k0, int K, int N, int tid)
{
    // ---- A tile: 128 rows x 32 halves, 2 threads/row, 2x b128 DMA each
    {
        const int row = tid >> 1;
        const int kh  = (tid & 1) * 16;
        const _Float16* src = Ab + (long long)(bm + row) * K + k0 + kh;
        _Float16* dst = &lA[row * LDSS + kh];
        async_cp16B(src,     dst);
        async_cp16B(src + 8, dst + 8);
    }
    // ---- B tile
    if (BLAYOUT == 0) {                  // f32 weights: convert in-flight
        const int row = tid >> 1;
        const int kh  = (tid & 1) * 16;
        const float* src = Bf + (long long)(bn + row) * K + k0 + kh;
        __builtin_prefetch(src + BK, 0, 0);
        Frag t;
        #pragma unroll
        for (int j = 0; j < 16; ++j) t.h[j] = (_Float16)src[j];
        *(u32x4*)&lB[row * LDSS + kh]     = t.q[0];
        *(u32x4*)&lB[row * LDSS + kh + 8] = t.q[1];
    } else if (BLAYOUT == 1) {           // f16 [N,K]: async DMA
        const int row = tid >> 1;
        const int kh  = (tid & 1) * 16;
        const _Float16* src = Bh + (long long)(bn + row) * K + k0 + kh;
        _Float16* dst = &lB[row * LDSS + kh];
        async_cp16B(src,     dst);
        async_cp16B(src + 8, dst + 8);
    } else {                             // f16 [K,N]: transpose into LDS
        const int kr = tid >> 3;         // 0..31
        const int nh = (tid & 7) * 16;   // 0..112
        const _Float16* src = Bh + (long long)(k0 + kr) * N + bn + nh;
        Frag t;
        t.q[0] = *(const u32x4*)(src);
        t.q[1] = *(const u32x4*)(src + 8);
        #pragma unroll
        for (int j = 0; j < 16; ++j) lB[(nh + j) * LDSS + kr] = t.h[j];
    }
}

// ---------------------------------------------------------------------------
// WMMA GEMM:  C[M,N] = epilogue(alpha * A[M,K] x B)   (wave32, gfx1250)
//   EPI 0: f32 out = val
//   EPI 1: f16 out = val*val*0.1 + val*0.1        (quadratic activation)
//   EPI 2: f16 out = val
//   EPI 3: f32 out = (Res*0.5 + val*0.5)*0.1*sfin (residual + rescale)
// Block 128x128, 256 threads = 8 waves (2 along M x 4 along N),
// each wave: 4x2 tiles of v_wmma_f32_16x16x32_f16.
// Double-buffered LDS; next tile DMA'd (ASYNCcnt) while WMMAs consume current.
// All dims are multiples of 128 (M,N) and 32 (K): no divergence, EXEC all-1s.
// ---------------------------------------------------------------------------
template<int BLAYOUT, int EPI>
__global__ __launch_bounds__(256) void wmma_gemm_kernel(
    const _Float16* __restrict__ A, const void* __restrict__ Bv,
    void* __restrict__ Cv, const float* __restrict__ Res,
    int M, int N, int K,
    long long sAb, long long sBb, long long sCb,
    float alpha, float sfin)
{
    __shared__ _Float16 lA[2][BM * LDSS];
    __shared__ _Float16 lB[2][BN * LDSS];

    const int tid   = threadIdx.x;
    const int lane  = tid & 31;
    const int wave  = tid >> 5;
    const int wm    = wave & 1;    // 0..1  -> 64 rows
    const int wn    = wave >> 1;   // 0..3  -> 32 cols
    const int bm    = blockIdx.y * BM;
    const int bn    = blockIdx.x * BN;
    const long long z = blockIdx.z;

    const _Float16* Ab = A + z * sAb;
    const float*    Bf = (const float*)Bv    + z * sBb;
    const _Float16* Bh = (const _Float16*)Bv + z * sBb;

    v8f acc[4][2] = {};

    const int laneq = lane & 15;
    const int kb    = (lane >> 4) * 8;   // 0 or 8 per ISA A/B f16 layout

    // prologue: stage tile 0 into buffer 0
    stage_tile<BLAYOUT>(Ab, Bf, Bh, lA[0], lB[0], bm, bn, 0, K, N, tid);
    wait_async0();
    __syncthreads();

    int p = 0;
    for (int k0 = 0; k0 < K; k0 += BK) {
        // prefetch next tile into the other buffer (uniform branch)
        if (k0 + BK < K)
            stage_tile<BLAYOUT>(Ab, Bf, Bh, lA[p ^ 1], lB[p ^ 1],
                                bm, bn, k0 + BK, K, N, tid);

        // ---- fragments per ISA f16 layout, two ds_load_b128 each
        const _Float16* lAp = lA[p];
        const _Float16* lBp = lB[p];
        Frag aF[4], bF[2];
        #pragma unroll
        for (int i = 0; i < 4; ++i) {
            const int r = wm * 64 + i * 16 + laneq;
            aF[i].q[0] = *(const u32x4*)&lAp[r * LDSS + kb];
            aF[i].q[1] = *(const u32x4*)&lAp[r * LDSS + kb + 16];
        }
        #pragma unroll
        for (int j = 0; j < 2; ++j) {
            const int c = wn * 32 + j * 16 + laneq;
            bF[j].q[0] = *(const u32x4*)&lBp[c * LDSS + kb];
            bF[j].q[1] = *(const u32x4*)&lBp[c * LDSS + kb + 16];
        }
        #pragma unroll
        for (int i = 0; i < 4; ++i)
            #pragma unroll
            for (int j = 0; j < 2; ++j)
                acc[i][j] = __builtin_amdgcn_wmma_f32_16x16x32_f16(
                    false, aF[i].v, false, bF[j].v, (short)0, acc[i][j],
                    false, false);

        wait_async0();      // own DMA into buffer p^1 landed
        __syncthreads();    // everyone done reading p / writing p^1
        p ^= 1;
    }

    // ---- epilogue (C/D layout: VGPR vi -> M = vi (+8 for lanes 16..31))
    const int rowOff = (lane >> 4) << 3;
    #pragma unroll
    for (int i = 0; i < 4; ++i) {
        #pragma unroll
        for (int j = 0; j < 2; ++j) {
            #pragma unroll
            for (int vi = 0; vi < 8; ++vi) {
                const int row = bm + wm * 64 + i * 16 + rowOff + vi;
                const int col = bn + wn * 32 + j * 16 + laneq;
                const long long ci = z * sCb + (long long)row * N + col;
                const float val = acc[i][j][vi] * alpha;
                if (EPI == 0) {
                    ((float*)Cv)[ci] = val;
                } else if (EPI == 1) {
                    const float y = val * val * 0.1f + val * 0.1f;
                    ((_Float16*)Cv)[ci] = (_Float16)y;
                } else if (EPI == 2) {
                    ((_Float16*)Cv)[ci] = (_Float16)val;
                } else {
                    const float r = Res[ci];
                    ((float*)Cv)[ci] = (r * 0.5f + val * 0.5f) * 0.1f * sfin;
                }
            }
        }
    }
}

// ---------------------------------------------------------------------------
// Pointwise kernels
// ---------------------------------------------------------------------------
__global__ void embed_kernel(const int* __restrict__ idx,
                             const float* __restrict__ wte,
                             const float* __restrict__ wpe,
                             float* __restrict__ x,
                             int T, int E, long long total)
{
    const long long i = (long long)blockIdx.x * blockDim.x + threadIdx.x;
    if (i >= total) return;
    const int e  = (int)(i & (E - 1));        // E = 1024
    const long long bt = i >> 10;
    const int t  = (int)(bt & (T - 1));       // T = 2048
    const int tok = idx[bt];
    x[i] = wte[(long long)tok * E + e] * 0.01f + wpe[(long long)t * E + e] * 0.01f;
}

// h16 = (f16)(x * w[e] * 0.1)   (FHELayerNorm = per-feature scale)
__global__ void scale_f16_kernel(const float* __restrict__ x,
                                 const float* __restrict__ w,
                                 _Float16* __restrict__ out,
                                 long long total, int emask)
{
    const long long i = (long long)blockIdx.x * blockDim.x + threadIdx.x;
    if (i >= total) return;
    out[i] = (_Float16)(x[i] * w[i & emask] * 0.1f);
}

// ---------------------------------------------------------------------------
// Orchestration
// ---------------------------------------------------------------------------
#define GEMM(BL, EP, Aptr, Bptr, Cptr, Rptr, M_, N_, K_, NB, sA, sB, sC, al, sf) \
    wmma_gemm_kernel<BL, EP><<<dim3((N_) / BN, (M_) / BM, (NB)), 256, 0, stream>>>( \
        (const _Float16*)(Aptr), (const void*)(Bptr), (void*)(Cptr),               \
        (const float*)(Rptr), (M_), (N_), (K_),                                    \
        (long long)(sA), (long long)(sB), (long long)(sC), (al), (sf))

extern "C" void kernel_launch(void* const* d_in, const int* in_sizes, int n_in,
                              void* d_out, int out_size, void* d_ws, size_t ws_size,
                              hipStream_t stream)
{
    (void)in_sizes; (void)n_in; (void)out_size; (void)ws_size;

    const int Bb = 2, T = 2048, E = 1024, V = 32000, L = 12;
    const int MT = Bb * T;                       // 4096 rows
    const long long TE = (long long)T * E;
    const long long TT = (long long)T * T;

    const int*   idx = (const int*)  d_in[0];
    const float* wte = (const float*)d_in[1];
    const float* wpe = (const float*)d_in[2];
    const float* ln1 = (const float*)d_in[3];
    const float* Wq  = (const float*)d_in[4];
    const float* Wk  = (const float*)d_in[5];
    const float* Wv  = (const float*)d_in[6];
    const float* Wo  = (const float*)d_in[7];
    const float* ln2 = (const float*)d_in[8];
    const float* f1  = (const float*)d_in[9];
    const float* f2  = (const float*)d_in[10];
    const float* lnf = (const float*)d_in[11];
    const float* lmh = (const float*)d_in[12];
    float* logits = (float*)d_out;

    // workspace carve-up (~88 MB)
    char* ws = (char*)d_ws;
    size_t off = 0;
    auto carve = [&](size_t bytes) {
        void* p = ws + off;
        off += (bytes + 255) & ~(size_t)255;
        return p;
    };
    float*    x    = (float*)   carve((size_t)MT * E * 4);      // f32 residual stream
    _Float16* h16  = (_Float16*)carve((size_t)MT * E * 2);      // normed acts
    _Float16* q16  = (_Float16*)carve((size_t)MT * E * 2);
    _Float16* k16  = (_Float16*)carve((size_t)MT * E * 2);
    _Float16* v16  = (_Float16*)carve((size_t)MT * E * 2);
    _Float16* a16  = (_Float16*)carve((size_t)Bb * T * T * 2);  // attn scores
    _Float16* ao16 = (_Float16*)carve((size_t)MT * E * 2);
    _Float16* m16  = (_Float16*)carve((size_t)MT * 2 * E * 2);  // MLP hidden

    const long long nXE = (long long)MT * E;
    const int EW = 256;

    // x = wte[idx]*0.01 + wpe[t]*0.01
    embed_kernel<<<(unsigned)((nXE + EW - 1) / EW), EW, 0, stream>>>(
        idx, wte, wpe, x, T, E, nXE);

    for (int l = 0; l < L; ++l) {
        const float* Wq_l = Wq + (size_t)l * E * E;
        const float* Wk_l = Wk + (size_t)l * E * E;
        const float* Wv_l = Wv + (size_t)l * E * E;
        const float* Wo_l = Wo + (size_t)l * E * E;
        const float* f1_l = f1 + (size_t)l * 2 * E * E;
        const float* f2_l = f2 + (size_t)l * 2 * E * E;
        const float  s    = (l < L - 1) ? 0.1f : 1.0f;

        // h = x * ln1 * 0.1  -> f16
        scale_f16_kernel<<<(unsigned)((nXE + EW - 1) / EW), EW, 0, stream>>>(
            x, ln1 + (size_t)l * E, h16, nXE, E - 1);

        // q/k/v projections ('bte,fe->btf': B = W[N,K], K-contiguous, f32)
        GEMM(0, 2, h16, Wq_l, q16, nullptr, MT, E, E, 1, 0, 0, 0, 0.01f, 1.0f);
        GEMM(0, 2, h16, Wk_l, k16, nullptr, MT, E, E, 1, 0, 0, 0, 0.01f, 1.0f);
        GEMM(0, 2, h16, Wv_l, v16, nullptr, MT, E, E, 1, 0, 0, 0, 0.1f,  1.0f);

        // a = (q k^T)*0.01, then a*a*0.1 + a*0.1   (batched, B = k16 [N,K] f16)
        GEMM(1, 1, q16, k16, a16, nullptr, T, T, E, Bb, TE, TE, TT, 0.01f, 1.0f);

        // ao = (a v)*0.01                          (batched, B = v16 [K,N] f16)
        GEMM(2, 2, a16, v16, ao16, nullptr, T, E, T, Bb, TT, TE, TE, 0.01f, 1.0f);

        // attn_out = (ao Wo)*0.1 ; x = (x*0.5 + attn*0.5)*0.1
        GEMM(0, 3, ao16, Wo_l, x, x, MT, E, E, 1, 0, 0, 0, 0.1f, 1.0f);

        // h2 = x * ln2 * 0.1 -> f16
        scale_f16_kernel<<<(unsigned)((nXE + EW - 1) / EW), EW, 0, stream>>>(
            x, ln2 + (size_t)l * E, h16, nXE, E - 1);

        // m = (h2 fc1)*0.05, quad activation -> f16 [MT, 2E]
        GEMM(0, 1, h16, f1_l, m16, nullptr, MT, 2 * E, E, 1, 0, 0, 0, 0.05f, 1.0f);

        // m2 = (m fc2)*0.05 ; x = ((x*0.5 + m2*0.5)*0.1)*s
        GEMM(0, 3, m16, f2_l, x, x, MT, E, 2 * E, 1, 0, 0, 0, 0.05f, s);
    }

    // final norm + lm_head: logits = (x*lnf*0.1) @ lm_head^T * 0.5  (f32 out)
    scale_f16_kernel<<<(unsigned)((nXE + EW - 1) / EW), EW, 0, stream>>>(
        x, lnf, h16, nXE, E - 1);
    GEMM(0, 0, h16, lmh, logits, nullptr, MT, V, E, 1, 0, 0, 0, 0.5f, 1.0f);
}
